// GATSCFC_77498389889094
// MI455X (gfx1250) — matrix operand order repaired and checked by reading, not compiled, not run on hardware
//
#include <hip/hip_runtime.h>
#include <math.h>

// ---------------- problem constants ----------------
#define NN     8192
#define EE     262144
#define HIDD   256
#define HEADS  4
#define EP     (EE + NN)       // edges + self loops = 270336
#define LAYERS 3
#define NEG_SLOPE 0.2f
#define LN_EPS 1e-5f

// ---------------- WMMA types ----------------
typedef __attribute__((ext_vector_type(16))) __bf16 bf16x16;
typedef __attribute__((ext_vector_type(8)))  __bf16 bf16x8;
typedef __attribute__((ext_vector_type(8)))  float  f32x8;

#define AS1 __attribute__((address_space(1)))
#define AS3 __attribute__((address_space(3)))

typedef int v4i_vs __attribute__((vector_size(16)));   // matches builtin param type

// ---- async global->LDS copy (ASYNCcnt path); builtin if present, else asm ----
__device__ __forceinline__ void async_copy_b128(const void* g, void* l) {
#if __has_builtin(__builtin_amdgcn_global_load_async_to_lds_b128)
    __builtin_amdgcn_global_load_async_to_lds_b128(
        (AS1 v4i_vs*)(uintptr_t)g, (AS3 v4i_vs*)(uintptr_t)l, 0, 0);
#else
    unsigned loff = (unsigned)(uintptr_t)l;   // AS3 offset = low 32 bits
    asm volatile("global_load_async_to_lds_b128 %0, %1, off"
                 :: "v"(loff), "v"(g) : "memory");
#endif
}

#if __has_builtin(__builtin_amdgcn_s_wait_asynccnt)
#define WAIT_ASYNC(n) __builtin_amdgcn_s_wait_asynccnt(n)
#else
#define WAIT_ASYNC(n) asm volatile("s_wait_asynccnt %0" :: "i"(n) : "memory")
#endif

// ---------------- weight prep: f32 K x N  ->  bf16 N x K (transposed) -------
__global__ __launch_bounds__(256) void convert_transpose_bf16(
    const float* __restrict__ W, __bf16* __restrict__ BT, int K, int Nn)
{
    __shared__ float t[32][33];
    const int k0 = blockIdx.x * 32, n0 = blockIdx.y * 32;
    const int tx = threadIdx.x & 31, ty = threadIdx.x >> 5;   // ty 0..7
#pragma unroll
    for (int i = ty; i < 32; i += 8)
        t[i][tx] = W[(size_t)(k0 + i) * Nn + n0 + tx];
    __syncthreads();
#pragma unroll
    for (int i = ty; i < 32; i += 8)
        BT[(size_t)(n0 + i) * K + k0 + tx] = (__bf16)t[tx][i];
}

// ---------------- GEMM: C = A(f32,MxK) @ B + bias, optional ELU -------------
// A: f32 row-major, converted to bf16 during LDS staging (v_cvt_pk_bf16_f32).
// B: pre-converted bf16, stored transposed (Nn x K row-major) -> async b128
//    copies straight into the [n][k] LDS layout the WMMA fragments want.
#define BM 64
#define BN 64
#define BK 32
#define LPA 40   // LDS pitch (bf16 elems): 80B rows, 16B-aligned frags, no bank conflicts
#define LPB 40

template<bool DO_ELU>
__global__ __launch_bounds__(256) void gemm_bf16_wmma(
    const float* __restrict__ A, const __bf16* __restrict__ BT,
    const float* __restrict__ bias, float* __restrict__ C,
    int M, int Nn, int K)
{
    __shared__ __bf16 As[BM * LPA];
    __shared__ __bf16 Bs[2 * BN * LPB];     // double buffered

    const int tid  = threadIdx.x;
    const int lane = tid & 31;
    const int wave = tid >> 5;               // 0..7
    const long bm = (long)blockIdx.x * BM;
    const long bn = (long)blockIdx.y * BN;

    // 4 (M) x 2 (N) wave grid; each wave: 16 rows x 32 cols = two 16x16 tiles
    const int wm = (wave & 3) * 16;
    const int wn = (wave >> 2) * 32;
    const int half = lane >> 4;              // K-half selector
    const int l15  = lane & 15;              // M row (A frag) / N col (B frag)

    // B staging: 1 async b128 per thread per tile (64 rows x 4 chunks)
    const int sn = tid >> 2;                 // 0..63
    const int sk = (tid & 3) * 8;            // bf16 elems: 0,8,16,24
    const __bf16* gB = BT + (size_t)(bn + sn) * K + sk;

    // A staging: 2 float4 per thread per tile
    const int ar = tid >> 3;                 // 0..31
    const int ac = (tid & 7) * 4;

    f32x8 acc0 = {};
    f32x8 acc1 = {};

    const int T = K / BK;
    // prologue: async-load B tile 0
    async_copy_b128(gB, &Bs[sn * LPB + sk]);

    for (int t = 0; t < T; ++t) {
        const int k0 = t * BK;
        const __bf16* curB = &Bs[(t & 1) * BN * LPB];

        // ---- stage A tile 64x32: f32 -> bf16 ----
#pragma unroll
        for (int rr = 0; rr < 2; ++rr) {
            const int row = ar + rr * 32;
            const float4 v = *(const float4*)(A + (size_t)(bm + row) * K + k0 + ac);
            __bf16* d = &As[row * LPA + ac];
            d[0] = (__bf16)v.x; d[1] = (__bf16)v.y;
            d[2] = (__bf16)v.z; d[3] = (__bf16)v.w;
        }

        // ---- issue async for next B tile, wait for current ----
        if (t + 1 < T) {
            async_copy_b128(gB + (t + 1) * BK,
                            &Bs[((t + 1) & 1) * BN * LPB + sn * LPB + sk]);
            __builtin_prefetch(A + (size_t)(bm + ar) * K + k0 + BK, 0, 1);
            WAIT_ASYNC(1);      // tile t landed; tile t+1 may stay in flight
        } else {
            WAIT_ASYNC(0);
        }
        __syncthreads();

        // ---- A fragment (ISA 16-bit 16x32 layout) ----
        const __bf16* ap = &As[(wm + l15) * LPA + half * 8];
        bf16x8 a0 = *(const bf16x8*)(ap);
        bf16x8 a1 = *(const bf16x8*)(ap + 16);
        bf16x16 af;
#pragma unroll
        for (int i = 0; i < 8; ++i) { af[i] = a0[i]; af[8 + i] = a1[i]; }

        // ---- B fragments: lane n = l15 (+16 for 2nd tile), K-chunks = half*8, 16+half*8 ----
        const __bf16* bp0 = &curB[(wn      + l15) * LPB + half * 8];
        const __bf16* bp1 = &curB[(wn + 16 + l15) * LPB + half * 8];
        bf16x8 b00 = *(const bf16x8*)(bp0), b01 = *(const bf16x8*)(bp0 + 16);
        bf16x8 b10 = *(const bf16x8*)(bp1), b11 = *(const bf16x8*)(bp1 + 16);
        bf16x16 bf0, bf1;
#pragma unroll
        for (int i = 0; i < 8; ++i) {
            bf0[i] = b00[i]; bf0[8 + i] = b01[i];
            bf1[i] = b10[i]; bf1[8 + i] = b11[i];
        }

        acc0 = __builtin_amdgcn_wmma_f32_16x16x32_bf16(false, af, false, bf0,
                                                       (short)0, acc0, false, false);
        acc1 = __builtin_amdgcn_wmma_f32_16x16x32_bf16(false, af, false, bf1,
                                                       (short)0, acc1, false, false);
        __syncthreads();
    }

    // ---- epilogue: C/D layout lane<16 -> M=r, lane>=16 -> M=r+8; N=l15 ----
    const long n0 = bn + wn + l15;
    const long n1 = n0 + 16;
    const float bia0 = bias[n0];
    const float bia1 = bias[n1];
#pragma unroll
    for (int r = 0; r < 8; ++r) {
        const long m = bm + wm + r + half * 8;
        float v0 = acc0[r] + bia0;
        float v1 = acc1[r] + bia1;
        if (DO_ELU) {
            v0 = v0 > 0.f ? v0 : (expf(v0) - 1.f);
            v1 = v1 > 0.f ? v1 : (expf(v1) - 1.f);
        }
        C[(size_t)m * Nn + n0] = v0;
        C[(size_t)m * Nn + n1] = v1;
    }
}

// ---------------- small utility kernels ----------------
__global__ __launch_bounds__(256) void fill_kernel(float* p, float v, int n) {
    int i = blockIdx.x * 256 + threadIdx.x;
    if (i < n) p[i] = v;
}

__global__ __launch_bounds__(256) void copy_kernel(const float* __restrict__ s,
                                                   float* __restrict__ d, int n) {
    int i = blockIdx.x * 256 + threadIdx.x;
    if (i < n) d[i] = s[i];
}

__device__ inline void atomicMaxF(float* addr, float val) {
    unsigned int* ua = (unsigned int*)addr;
    unsigned int old = __float_as_uint(*addr);
    while (__uint_as_float(old) < val) {
        unsigned int assumed = old;
        old = atomicCAS(ua, assumed, __float_as_uint(val));
        if (old == assumed) break;
    }
}

// ---------------- self-loop edge_attr (PyG fill_value='mean') ---------------
__global__ __launch_bounds__(256) void edge_count_kernel(
    const int* __restrict__ ei, const float* __restrict__ eattr,
    float* cnt, float* asum)
{
    int e = blockIdx.x * 256 + threadIdx.x;
    if (e >= EE) return;
    int d = ei[EE + e];
    atomicAdd(&cnt[d], 1.0f);
    atomicAdd(&asum[d], eattr[e]);
}

__global__ __launch_bounds__(256) void loop_attr_kernel(
    const float* __restrict__ cnt, const float* __restrict__ asum, float* lattr)
{
    int i = blockIdx.x * 256 + threadIdx.x;
    if (i < NN) lattr[i] = asum[i] / fmaxf(cnt[i], 1.0f);
}

// ---------------- GATv2 edge logits + segment max (1 wave / edge) -----------
__global__ __launch_bounds__(256) void edge_logits_kernel(
    const float* __restrict__ xl, const float* __restrict__ xr,
    const int* __restrict__ ei, const float* __restrict__ eattr,
    const float* __restrict__ lattr, const float* __restrict__ We,
    const float* __restrict__ attv, float* __restrict__ logits,
    float* __restrict__ lmax)
{
    const int wid  = blockIdx.x * 8 + (threadIdx.x >> 5);
    const int lane = threadIdx.x & 31;
    if (wid >= EP) return;
    int src, dst; float ea;
    if (wid < EE) { src = ei[wid]; dst = ei[EE + wid]; ea = eattr[wid]; }
    else          { src = dst = wid - EE; ea = lattr[src]; }

    const int j0 = lane * 8;                 // 8 channels per lane, same head
    const float* pl = xl + (size_t)src * HIDD + j0;
    const float* pr = xr + (size_t)dst * HIDD + j0;
    float part = 0.f;
#pragma unroll
    for (int i = 0; i < 8; ++i) {
        float m = pl[i] + pr[i] + ea * We[j0 + i];
        m = m > 0.f ? m : NEG_SLOPE * m;
        part += m * attv[j0 + i];
    }
    part += __shfl_xor(part, 1);
    part += __shfl_xor(part, 2);
    part += __shfl_xor(part, 4);
    if ((lane & 7) == 0) {
        int h = lane >> 3;
        logits[(size_t)wid * HEADS + h] = part;
        atomicMaxF(&lmax[dst * HEADS + h], part);
    }
}

// ---------------- exp + segment sum ----------------
__global__ __launch_bounds__(256) void edge_exp_kernel(
    const int* __restrict__ ei, float* __restrict__ logits,
    const float* __restrict__ lmax, float* __restrict__ denom)
{
    int t = blockIdx.x * 256 + threadIdx.x;          // t < EP*HEADS
    if (t >= EP * HEADS) return;
    int e = t >> 2, h = t & 3;
    int dst = (e < EE) ? ei[EE + e] : (e - EE);
    float ex = expf(logits[t] - lmax[dst * HEADS + h]);
    logits[t] = ex;
    atomicAdd(&denom[dst * HEADS + h], ex);
}

// ---------------- weighted scatter-sum (1 wave / edge) ----------------------
__global__ __launch_bounds__(256) void edge_scatter_kernel(
    const float* __restrict__ xl, const int* __restrict__ ei,
    const float* __restrict__ ex, const float* __restrict__ denom,
    float* __restrict__ acc)
{
    const int wid  = blockIdx.x * 8 + (threadIdx.x >> 5);
    const int lane = threadIdx.x & 31;
    if (wid >= EP) return;
    int src, dst;
    if (wid < EE) { src = ei[wid]; dst = ei[EE + wid]; }
    else          { src = dst = wid - EE; }
    const int h  = lane >> 3;
    const float alpha = ex[(size_t)wid * HEADS + h] / denom[dst * HEADS + h];
    const int j0 = lane * 8;
    const float* pl = xl + (size_t)src * HIDD + j0;
    float* pa = acc + (size_t)dst * HIDD + j0;
#pragma unroll
    for (int i = 0; i < 8; ++i) atomicAdd(&pa[i], pl[i] * alpha);
}

// ---------------- +b_conv, ELU, +residual, LayerNorm (1 wave / row) ---------
__global__ __launch_bounds__(256) void post_ln_kernel(
    const float* __restrict__ acc, const float* __restrict__ b_conv,
    const float* __restrict__ g, const float* __restrict__ b,
    float* __restrict__ h)
{
    const int row  = blockIdx.x * 8 + (threadIdx.x >> 5);
    const int lane = threadIdx.x & 31;
    const int c0   = lane * 8;
    const float* pa = acc + (size_t)row * HIDD + c0;
    float* ph = h + (size_t)row * HIDD + c0;
    float v[8], s = 0.f, s2 = 0.f;
#pragma unroll
    for (int i = 0; i < 8; ++i) {
        float a = pa[i] + b_conv[c0 + i];
        a = a > 0.f ? a : (expf(a) - 1.f);
        a += ph[i];
        v[i] = a; s += a; s2 += a * a;
    }
#pragma unroll
    for (int off = 16; off; off >>= 1) {
        s  += __shfl_xor(s,  off);
        s2 += __shfl_xor(s2, off);
    }
    const float mu  = s * (1.0f / HIDD);
    const float var = s2 * (1.0f / HIDD) - mu * mu;
    const float inv = rsqrtf(var + LN_EPS);
#pragma unroll
    for (int i = 0; i < 8; ++i)
        ph[i] = (v[i] - mu) * inv * g[c0 + i] + b[c0 + i];
}

// ---------------- in-place symmetrize: each i<=j pair owned once ------------
__global__ __launch_bounds__(256) void symmetrize_kernel(float* fc) {
    const long i = blockIdx.y * 16 + threadIdx.y;
    const long j = blockIdx.x * 16 + threadIdx.x;
    if (j < i) return;
    float a = fc[(size_t)i * NN + j];
    float c = fc[(size_t)j * NN + i];
    float avg = 0.5f * (a + c);
    fc[(size_t)i * NN + j] = avg;
    fc[(size_t)j * NN + i] = avg;
}

// ---------------- host-side orchestration ----------------
extern "C" void kernel_launch(void* const* d_in, const int* in_sizes, int n_in,
                              void* d_out, int out_size, void* d_ws, size_t ws_size,
                              hipStream_t stream) {
    const float* x      = (const float*)d_in[0];
    const int*   ei     = (const int*)  d_in[1];
    const float* eattr  = (const float*)d_in[2];
    const float* w_in   = (const float*)d_in[3];
    const float* b_in   = (const float*)d_in[4];
    const float* Wl     = (const float*)d_in[5];
    const float* bl     = (const float*)d_in[6];
    const float* Wr     = (const float*)d_in[7];
    const float* br     = (const float*)d_in[8];
    const float* We     = (const float*)d_in[9];
    const float* att    = (const float*)d_in[10];
    const float* b_conv = (const float*)d_in[11];
    const float* ln_g   = (const float*)d_in[12];
    const float* ln_b   = (const float*)d_in[13];
    const float* w1     = (const float*)d_in[14];
    const float* b1     = (const float*)d_in[15];
    const float* w2     = (const float*)d_in[16];
    const float* b2     = (const float*)d_in[17];
    float* out = (float*)d_out;

    // workspace carve-up: ~38 MB f32 + ~9.3 MB bf16
    float* h      = (float*)d_ws;
    float* xl     = h      + (size_t)NN * HIDD;
    float* xr     = xl     + (size_t)NN * HIDD;
    float* acc    = xr     + (size_t)NN * HIDD;
    float* logits = acc    + (size_t)NN * HIDD;
    float* lmax   = logits + (size_t)EP * HEADS;
    float* denom  = lmax   + (size_t)NN * HEADS;
    float* cnt    = denom  + (size_t)NN * HEADS;
    float* asum   = cnt    + NN;
    float* lattr  = asum   + NN;

    __bf16* bt_win = (__bf16*)(lattr + NN);                 // 256 x 8192
    __bf16* bt_wl  = bt_win + (size_t)HIDD * NN;            // 3 x 256 x 256
    __bf16* bt_wr  = bt_wl + (size_t)LAYERS * HIDD * HIDD;
    __bf16* bt_w1  = bt_wr + (size_t)LAYERS * HIDD * HIDD;  // 256 x 256
    __bf16* bt_w2  = bt_w1 + (size_t)HIDD * HIDD;           // 8192 x 256

    const dim3 b256(256);

    // ---- weight prep: convert+transpose to bf16 [n][k] ----
    convert_transpose_bf16<<<dim3(NN / 32, HIDD / 32), b256, 0, stream>>>(
        w_in, bt_win, NN, HIDD);
    for (int l = 0; l < LAYERS; ++l) {
        convert_transpose_bf16<<<dim3(HIDD / 32, HIDD / 32), b256, 0, stream>>>(
            Wl + (size_t)l * HIDD * HIDD, bt_wl + (size_t)l * HIDD * HIDD, HIDD, HIDD);
        convert_transpose_bf16<<<dim3(HIDD / 32, HIDD / 32), b256, 0, stream>>>(
            Wr + (size_t)l * HIDD * HIDD, bt_wr + (size_t)l * HIDD * HIDD, HIDD, HIDD);
    }
    convert_transpose_bf16<<<dim3(HIDD / 32, HIDD / 32), b256, 0, stream>>>(
        w1, bt_w1, HIDD, HIDD);
    convert_transpose_bf16<<<dim3(HIDD / 32, NN / 32), b256, 0, stream>>>(
        w2, bt_w2, HIDD, NN);

    // ---- self-loop attrs ----
    fill_kernel<<<dim3(NN / 256), b256, 0, stream>>>(cnt, 0.f, NN);
    fill_kernel<<<dim3(NN / 256), b256, 0, stream>>>(asum, 0.f, NN);
    edge_count_kernel<<<dim3(EE / 256), b256, 0, stream>>>(ei, eattr, cnt, asum);
    loop_attr_kernel<<<dim3(NN / 256), b256, 0, stream>>>(cnt, asum, lattr);

    // ---- input projection: h = ELU(x @ w_in + b_in) ----
    gemm_bf16_wmma<true><<<dim3(NN / BM, HIDD / BN), b256, 0, stream>>>(
        x, bt_win, b_in, h, NN, HIDD, NN);

    for (int l = 0; l < LAYERS; ++l) {
        gemm_bf16_wmma<false><<<dim3(NN / BM, HIDD / BN), b256, 0, stream>>>(
            h, bt_wl + (size_t)l * HIDD * HIDD, bl + l * HIDD, xl, NN, HIDD, HIDD);
        gemm_bf16_wmma<false><<<dim3(NN / BM, HIDD / BN), b256, 0, stream>>>(
            h, bt_wr + (size_t)l * HIDD * HIDD, br + l * HIDD, xr, NN, HIDD, HIDD);

        fill_kernel<<<dim3(NN * HEADS / 256), b256, 0, stream>>>(lmax, -3.0e38f, NN * HEADS);
        fill_kernel<<<dim3(NN * HEADS / 256), b256, 0, stream>>>(denom, 0.f, NN * HEADS);
        fill_kernel<<<dim3(NN * HIDD / 256), b256, 0, stream>>>(acc, 0.f, NN * HIDD);

        edge_logits_kernel<<<dim3(EP / 8), b256, 0, stream>>>(
            xl, xr, ei, eattr, lattr, We + l * HIDD, att + l * HIDD, logits, lmax);
        edge_exp_kernel<<<dim3(EP * HEADS / 256), b256, 0, stream>>>(ei, logits, lmax, denom);
        edge_scatter_kernel<<<dim3(EP / 8), b256, 0, stream>>>(xl, ei, logits, denom, acc);

        post_ln_kernel<<<dim3(NN / 8), b256, 0, stream>>>(
            acc, b_conv + l * HIDD, ln_g + l * HIDD, ln_b + l * HIDD, h);
    }

    // emb output
    copy_kernel<<<dim3(NN * HIDD / 256), b256, 0, stream>>>(h, out + (size_t)NN * NN, NN * HIDD);

    // fc = (elu(h@w1+b1) @ w2 + b2), then symmetrize in place
    gemm_bf16_wmma<true><<<dim3(NN / BM, HIDD / BN), b256, 0, stream>>>(
        h, bt_w1, b1, acc, NN, HIDD, HIDD);
    gemm_bf16_wmma<false><<<dim3(NN / BM, NN / BN), b256, 0, stream>>>(
        acc, bt_w2, b2, out, NN, NN, HIDD);
    symmetrize_kernel<<<dim3(NN / 16, NN / 16), dim3(16, 16), 0, stream>>>(out);
}
